// MetaPTP_69715909148802
// MI455X (gfx1250) — matrix operand order repaired and checked by reading, not compiled, not run on hardware
//
#include <hip/hip_runtime.h>
#include <hip/hip_bf16.h>

// ---------------------------------------------------------------------------
// MetaPTP forward for MI455X (gfx1250).  Heavy MLPs run as bf16 WMMA GEMMs
// (v_wmma_f32_16x16x32_bf16) with fragment-packed weights, GRU scan is
// launch-chained per step, small elementwise stages are VALU kernels.
// ---------------------------------------------------------------------------

typedef __attribute__((ext_vector_type(16))) __bf16 v16bf;
typedef __attribute__((ext_vector_type(8)))  float  v8f;

#define LL   8
#define LM1  7
#define NA   2048
#define NN   32
#define HID  256
#define MBIG (LM1*NA*NN)   /* 458752 */
#define MS   (LM1*NA)      /* 14336  */
#define MFX  (NA*NN)       /* 65536  */

// ---- bf16 <-> f32 helpers (bit-exact, no reliance on __bf16 arithmetic) ----
__device__ __forceinline__ __bf16 f2b(float f) {
    unsigned u = __builtin_bit_cast(unsigned, f);
    u += 0x7fffu + ((u >> 16) & 1u);            // round-to-nearest-even
    unsigned short h = (unsigned short)(u >> 16);
    return __builtin_bit_cast(__bf16, h);
}
__device__ __forceinline__ float b2f(__bf16 b) {
    unsigned short h = __builtin_bit_cast(unsigned short, b);
    unsigned u = ((unsigned)h) << 16;
    return __builtin_bit_cast(float, u);
}
__device__ __forceinline__ float sigf(float x) { return 1.f / (1.f + __expf(-x)); }

// ---------------------------------------------------------------------------
// Pack weight f32[K,N] -> bf16 in WMMA B-fragment order:
//   Wp[((ntile*(K/32) + kc)*32 + lane)*16 + e]  =  W[kc*32 + 16*(lane>=16) + e,
//                                                    ntile*16 + (lane&15)]
// so each lane's 16-element fragment is 32 contiguous bytes and a wave's
// B-tile is 1KB contiguous, streamed linearly over kc.
// ---------------------------------------------------------------------------
__global__ void pack_w(const float* __restrict__ Wf, __bf16* __restrict__ Wp, int K, int N) {
    long long idx = (long long)blockIdx.x * blockDim.x + threadIdx.x;
    if (idx >= (long long)K * N) return;
    int e    = (int)(idx & 15);
    int lane = (int)((idx >> 4) & 31);
    long long rest = idx >> 9;
    int kchunks = K >> 5;
    int kc    = (int)(rest % kchunks);
    int ntile = (int)(rest / kchunks);
    int kk  = kc * 32 + ((lane & 16) ? 16 : 0) + e;
    int nn2 = ntile * 16 + (lane & 15);
    Wp[idx] = f2b(Wf[(size_t)kk * N + nn2]);
}

// ---------------------------------------------------------------------------
// Preprocess: v, a, dp, dv, dist, bearing, tau, mpd, masks, dp0
// ---------------------------------------------------------------------------
__global__ void preprocess(const float* __restrict__ x, const float* __restrict__ nb,
                           float* __restrict__ kfeat,  // (7,2048,32,3)
                           float* __restrict__ nfeat,  // (7,2048,32,4)
                           float* __restrict__ sfeat,  // (7,2048,4)
                           unsigned char* __restrict__ mask,   // (7,2048,32)
                           float* __restrict__ dp0,    // (2048,32,2)
                           unsigned char* __restrict__ mask0)  // (2048,32)
{
    int idx = blockIdx.x * blockDim.x + threadIdx.x;
    if (idx >= MBIG) return;
    int k = idx & 31;
    int n = (idx >> 5) & (NA - 1);
    int l = idx >> 16;                 // 2048*32 == 1<<16
    int t = l + 1;

    const float* xt  = x + ((size_t)t * NA + n) * 6;
    const float* xtm = x + ((size_t)(t - 1) * NA + n) * 6;
    float vx = xt[0] - xtm[0], vy = xt[1] - xtm[1];

    const float* nbp = nb + (((size_t)t * NA + n) * NN + k) * 4;
    float dpx = nbp[0] - xt[0], dpy = nbp[1] - xt[1];
    float dvx = nbp[2] - vx,    dvy = nbp[3] - vy;

    float dist = sqrtf(dpx * dpx + dpy * dpy);
    mask[idx] = (dist <= 2.0f) ? 1 : 0;

    float speed   = sqrtf(vx * vx + vy * vy);
    float bearing = (dpx * vx + dpy * vy) / (dist * speed);
    if (!__builtin_isfinite(bearing)) bearing = 0.f;

    float dvn = sqrtf(dvx * dvx + dvy * dvy);
    float tau = -(dpx * dvx + dpy * dvy) / dvn;
    if (!__builtin_isfinite(tau)) tau = 0.f;
    tau = fminf(fmaxf(tau, 0.f), 7.f);
    float mpx = dpx + tau * dvx, mpy = dpy + tau * dvy;
    float mpd = sqrtf(mpx * mpx + mpy * mpy);

    kfeat[(size_t)idx * 3 + 0] = dist;
    kfeat[(size_t)idx * 3 + 1] = bearing;
    kfeat[(size_t)idx * 3 + 2] = mpd;
    nfeat[(size_t)idx * 4 + 0] = dpx;
    nfeat[(size_t)idx * 4 + 1] = dpy;
    nfeat[(size_t)idx * 4 + 2] = dvx;
    nfeat[(size_t)idx * 4 + 3] = dvy;

    if (k == 0) {
        float ax, ay;
        if (l == 0) { ax = xt[4]; ay = xt[5]; }        // x[1,:,4:6]
        else {
            const float* xtm2 = x + ((size_t)(t - 2) * NA + n) * 6;
            ax = vx - (xtm[0] - xtm2[0]);
            ay = vy - (xtm[1] - xtm2[1]);
        }
        float* sp = sfeat + ((size_t)l * NA + n) * 4;
        sp[0] = vx; sp[1] = vy; sp[2] = ax; sp[3] = ay;
    }
    if (l == 0) {                                       // t=0 slice: dp0/mask0
        const float* x0  = x + (size_t)n * 6;
        const float* nb0 = nb + ((size_t)n * NN + k) * 4;
        float d0x = nb0[0] - x0[0], d0y = nb0[1] - x0[1];
        float dd  = sqrtf(d0x * d0x + d0y * d0y);
        int i0 = n * NN + k;
        dp0[(size_t)i0 * 2 + 0] = d0x;
        dp0[(size_t)i0 * 2 + 1] = d0y;
        mask0[i0] = (dd <= 2.0f) ? 1 : 0;
    }
}

// ---------------------------------------------------------------------------
// Tiny-K dense layer (K<=4): Y = relu6(X*W + b), f32 in, bf16 out
// ---------------------------------------------------------------------------
__global__ void dense_small(const float* __restrict__ X, const float* __restrict__ Wf,
                            const float* __restrict__ bias, __bf16* __restrict__ Y,
                            long long M, int Kin, int Nout, int act)
{
    long long idx = (long long)blockIdx.x * blockDim.x + threadIdx.x;
    if (idx >= M * Nout) return;
    int  nn = (int)(idx % Nout);
    long long m = idx / Nout;
    float acc = bias[nn];
    const float* xp = X + m * Kin;
    for (int kk = 0; kk < Kin; ++kk) acc += xp[kk] * Wf[kk * Nout + nn];
    if (act) acc = fminf(fmaxf(acc, 0.f), 6.f);
    Y[(size_t)m * Nout + nn] = f2b(acc);
}

// ---------------------------------------------------------------------------
// WMMA GEMM:  Y = act(X[MxK] * Wp + b),  bf16 A/B, f32 accumulate.
// Wp is fragment-packed (see pack_w).  Block = 256 threads = 8 waves
// (2 row-groups x 4 col-groups), tile 32x64, K consumed 64 per barrier pair.
// A tile staged in LDS (stride 72 elems = 144B, keeps b128 alignment).
// M % 32 == 0, K % 64 == 0, N % 16 == 0.
// ---------------------------------------------------------------------------
__global__ __launch_bounds__(256) void wmma_gemm(
    const __bf16* __restrict__ X, const __bf16* __restrict__ Wp,
    const float* __restrict__ bias,
    __bf16* __restrict__ Yb, float* __restrict__ Yf,
    int M, int K, int N, int act)
{
    __shared__ __bf16 Xs[32 * 72];

    const int tid  = threadIdx.x;
    const int lane = tid & 31;
    const int wave = tid >> 5;
    const int rg   = wave >> 2;                   // 0..1 row group
    const int cg   = wave & 3;                    // 0..3 col group
    const int row0 = blockIdx.x * 32;
    const int n0   = blockIdx.y * 64 + cg * 16;
    const bool active = (n0 < N);
    const int  ntile  = active ? (n0 >> 4) : 0;   // inactive waves: harmless tile 0

    const int mA  = lane & 15;                    // A row within tile
    const int kbA = (lane & 16) ? 16 : 0;         // A K sub-base per lane half
    const int kchunks = K >> 5;

    // per-wave packed-B stream: 1KB per 32-K-chunk, linear
    const __bf16* wp = Wp + (size_t)ntile * kchunks * 512 + lane * 16;

    // cooperative X staging: 256 threads x 8 contiguous bf16 = 32x64 tile
    const int srr = tid >> 3;
    const int scc = (tid & 7) * 8;

    union U16 { uint4 q[2]; v16bf v; };
    v8f acc = {};

    for (int kc = 0; kc < K; kc += 64) {
        *(uint4*)(Xs + srr * 72 + scc) =
            *(const uint4*)(X + (size_t)(row0 + srr) * K + kc + scc);
        __syncthreads();

        __builtin_prefetch((const void*)(wp + 1024), 0, 1);   // next K-64 of B
        const __bf16* xrow = Xs + (rg * 16 + mA) * 72;
        #pragma unroll
        for (int s = 0; s < 2; ++s) {
            U16 a, b;
            a.q[0] = *(const uint4*)(xrow + s * 32 + kbA);
            a.q[1] = *(const uint4*)(xrow + s * 32 + kbA + 8);
            b.q[0] = *(const uint4*)(wp);
            b.q[1] = *(const uint4*)(wp + 8);
            wp += 512;
            acc = __builtin_amdgcn_wmma_f32_16x16x32_bf16(
                      false, a.v, false, b.v, (short)0, acc, false, false);
        }
        __syncthreads();
    }

    if (active) {
        const int nB = lane & 15;
        float bv = bias ? bias[n0 + nB] : 0.f;
        #pragma unroll
        for (int j = 0; j < 8; ++j) {
            int row = row0 + rg * 16 + j + ((lane & 16) ? 8 : 0);
            float v = acc[j] + bv;
            if (act == 1) v = fminf(fmaxf(v, 0.f), 6.f);
            size_t off = (size_t)row * N + n0 + nB;
            if (Yb) Yb[off] = f2b(v);
            if (Yf) Yf[off] = v;
        }
    }
}

// ---------------------------------------------------------------------------
// h0 = sum_k mask0 * fx(dp0)   (f32 + bf16 copies of h)
// ---------------------------------------------------------------------------
__global__ void masked_sum(const float* __restrict__ fxout, const unsigned char* __restrict__ mask0,
                           float* __restrict__ hf, __bf16* __restrict__ hb)
{
    int idx = blockIdx.x * blockDim.x + threadIdx.x;    // 2048*256
    if (idx >= NA * HID) return;
    int n = idx >> 8, d = idx & 255;
    float s = 0.f;
    for (int k = 0; k < NN; ++k)
        if (mask0[n * NN + k]) s += fxout[(((size_t)n * NN + k) * HID) + d];
    hf[idx] = s;
    hb[idx] = f2b(s);
}

// ---------------------------------------------------------------------------
// ksum[n,d] = sum_l (s_e @ Wk)[l,n,d] ;  meta_w[l,n] = -qe[l,n,:].ksum[n,:]/16
// ---------------------------------------------------------------------------
__global__ void ksum_kernel(const float* __restrict__ sewk, float* __restrict__ ks)
{
    int idx = blockIdx.x * blockDim.x + threadIdx.x;    // 2048*256
    if (idx >= NA * HID) return;
    int n = idx >> 8, d = idx & 255;
    float s = 0.f;
    for (int l = 0; l < LM1; ++l) s += sewk[((size_t)l * NA + n) * HID + d];
    ks[idx] = s;
}
__global__ void metaw_kernel(const float* __restrict__ qe, const float* __restrict__ ks,
                             float* __restrict__ mw)
{
    int idx = blockIdx.x * blockDim.x + threadIdx.x;    // 7*2048
    if (idx >= MS) return;
    int n = idx & (NA - 1);
    float s = 0.f;
    for (int d = 0; d < HID; ++d) s += qe[(size_t)idx * HID + d] * ks[(size_t)n * HID + d];
    mw[idx] = -s * 0.0625f;                             // 1/sqrt(256)
}

// ---------------------------------------------------------------------------
// Attention + xin assembly for GRU step t.  One wave per agent.
// ---------------------------------------------------------------------------
__global__ __launch_bounds__(256) void attn_kernel(
    int t, const __bf16* __restrict__ k_e, const float* __restrict__ qf,
    const unsigned char* __restrict__ mask, const __bf16* __restrict__ n_e,
    const float* __restrict__ se_f, const float* __restrict__ metaw,
    __bf16* __restrict__ xin)
{
    __shared__ float esh[8][NN];
    int lane = threadIdx.x & 31, w = threadIdx.x >> 5;
    int n = blockIdx.x * 8 + w;
    size_t base = (size_t)t * NA + n;

    const __bf16* kp = k_e + base * NN * HID;
    const float*  qp = qf + (size_t)n * HID;

    for (int k = 0; k < NN; ++k) {                       // e[k] = k_t[k,:].q
        float p = 0.f;
        #pragma unroll
        for (int j = 0; j < 8; ++j) {
            int d = lane + 32 * j;
            p += b2f(kp[k * HID + d]) * qp[d];
        }
        for (int off = 16; off; off >>= 1) p += __shfl_xor(p, off, 32);
        if (lane == 0) esh[w][k] = (p > 0.f) ? p : 0.2f * p;   // leaky
    }
    __syncthreads();

    float e = esh[w][lane];
    bool  m = mask[base * NN + lane] != 0;
    unsigned long long bal = __ballot(m);
    bool valid = (bal & 0xffffffffull) != 0ull;
    float em = m ? e : -3.4e38f;
    float mx = em;
    for (int off = 16; off; off >>= 1) mx = fmaxf(mx, __shfl_xor(mx, off, 32));
    float p = (valid && m) ? __expf(em - mx) : 0.f;
    float s = p;
    for (int off = 16; off; off >>= 1) s += __shfl_xor(s, off, 32);
    float att = (valid && s > 0.f) ? p / s : 0.f;
    __syncthreads();
    esh[w][lane] = att;
    __syncthreads();

    const __bf16* np_ = n_e + base * NN * 128;
    __bf16* xp = xin + (size_t)n * 384;
    #pragma unroll
    for (int j = 0; j < 4; ++j) {                        // agg over 128 dims
        int d = lane + 32 * j;
        float agg = 0.f;
        for (int k = 0; k < NN; ++k) agg += esh[w][k] * b2f(np_[k * 128 + d]);
        xp[d] = f2b(agg);
    }
    float wt = metaw[base];
    const float* sp = se_f + base * HID;
    #pragma unroll
    for (int j = 0; j < 8; ++j) {                        // -s_t * w
        int d = lane + 32 * j;
        xp[128 + d] = f2b(-sp[d] * wt);
    }
}

// ---------------------------------------------------------------------------
// GRU gate nonlinearity + state update
// ---------------------------------------------------------------------------
__global__ void gru_update(const float* __restrict__ gi, const float* __restrict__ gh,
                           float* __restrict__ hf, __bf16* __restrict__ hb)
{
    int idx = blockIdx.x * blockDim.x + threadIdx.x;     // 2048*256
    if (idx >= NA * HID) return;
    int n = idx >> 8, d = idx & 255;
    const float* gin = gi + (size_t)n * 768;
    const float* ghn = gh + (size_t)n * 768;
    float r  = sigf(gin[d]       + ghn[d]);
    float z  = sigf(gin[256 + d] + ghn[256 + d]);
    float nv = tanhf(gin[512 + d] + r * ghn[512 + d]);
    float h  = hf[idx];
    float hn = (1.f - z) * nv + z * h;
    hf[idx] = hn;
    hb[idx] = f2b(hn);
}

// ---------------------------------------------------------------------------
// pred cumsum over horizon + last position
// ---------------------------------------------------------------------------
__global__ void final_kernel(const float* __restrict__ pred, const float* __restrict__ x,
                             float* __restrict__ out)
{
    int idx = blockIdx.x * blockDim.x + threadIdx.x;     // 2048*20*2
    if (idx >= NA * 20 * 2) return;
    int n = idx / 40;
    int r = idx % 40;
    int j = r >> 1, c = r & 1;
    float posl = x[((size_t)7 * NA + n) * 6 + c];
    const float* pr = pred + (size_t)n * 480 + j * 24 + c;
    float acc = 0.f;
    for (int i = 0; i < 12; ++i) {
        acc += pr[i * 2];
        out[((((size_t)n * 20 + j) * 12) + i) * 2 + c] = acc + posl;
    }
}

// ---------------------------------------------------------------------------
// Host launcher
// ---------------------------------------------------------------------------
static inline dim3 grid1(long long total, int bs) { return dim3((unsigned)((total + bs - 1) / bs)); }

extern "C" void kernel_launch(void* const* d_in, const int* in_sizes, int n_in,
                              void* d_out, int out_size, void* d_ws, size_t ws_size,
                              hipStream_t stream)
{
    (void)in_sizes; (void)n_in; (void)out_size; (void)ws_size;

    // JAX pytree order: x, neighbor, then params with sorted keys:
    // Wk, Wq, embed_k(w,b)x3, embed_n(w,b)x3, embed_q(w,b)x3, embed_s(w,b)x2,
    // fx_init(w,b)x3, fy_init(w,b)x2, gru(Wih,Whh,bih,bhh), reg_head(w,b)x2
    const float* x    = (const float*)d_in[0];
    const float* nb   = (const float*)d_in[1];
    const float* Wk_f = (const float*)d_in[2];
    const float* Wq_f = (const float*)d_in[3];
    const float* ekw1 = (const float*)d_in[4];  const float* ekb1 = (const float*)d_in[5];
    const float* ekw2 = (const float*)d_in[6];  const float* ekb2 = (const float*)d_in[7];
    const float* ekw3 = (const float*)d_in[8];  const float* ekb3 = (const float*)d_in[9];
    const float* enw1 = (const float*)d_in[10]; const float* enb1 = (const float*)d_in[11];
    const float* enw2 = (const float*)d_in[12]; const float* enb2 = (const float*)d_in[13];
    const float* enw3 = (const float*)d_in[14]; const float* enb3 = (const float*)d_in[15];
    const float* eqw1 = (const float*)d_in[16]; const float* eqb1 = (const float*)d_in[17];
    const float* eqw2 = (const float*)d_in[18]; const float* eqb2 = (const float*)d_in[19];
    const float* eqw3 = (const float*)d_in[20]; const float* eqb3 = (const float*)d_in[21];
    const float* esw1 = (const float*)d_in[22]; const float* esb1 = (const float*)d_in[23];
    const float* esw2 = (const float*)d_in[24]; const float* esb2 = (const float*)d_in[25];
    const float* fxw1 = (const float*)d_in[26]; const float* fxb1 = (const float*)d_in[27];
    const float* fxw2 = (const float*)d_in[28]; const float* fxb2 = (const float*)d_in[29];
    const float* fxw3 = (const float*)d_in[30]; const float* fxb3 = (const float*)d_in[31];
    const float* fyw1 = (const float*)d_in[32]; const float* fyb1 = (const float*)d_in[33];
    const float* fyw2 = (const float*)d_in[34]; const float* fyb2 = (const float*)d_in[35];
    const float* Wih_f = (const float*)d_in[36];
    const float* Whh_f = (const float*)d_in[37];
    const float* bih   = (const float*)d_in[38];
    const float* bhh   = (const float*)d_in[39];
    const float* rgw1 = (const float*)d_in[40]; const float* rgb1 = (const float*)d_in[41];
    const float* rgw2 = (const float*)d_in[42]; const float* rgb2 = (const float*)d_in[43];

    // --- workspace bump allocator -------------------------------------------
    char* wsp = (char*)d_ws;
    size_t off = 0;
    auto alloc = [&](size_t bytes) -> void* {
        void* p = wsp + off;
        off = (off + bytes + 255) & ~(size_t)255;
        return p;
    };

    // packed bf16 weights for WMMA GEMMs
    struct WRec { const float* src; int K, N; __bf16* dst; };
    WRec wr[] = {
        {Wk_f, HID, HID, 0}, {Wq_f, HID, HID, 0},
        {ekw2, HID, HID, 0}, {ekw3, HID, HID, 0},
        {enw2, 64, 128, 0},  {enw3, 128, 128, 0},
        {eqw1, HID, HID, 0}, {eqw2, HID, HID, 0}, {eqw3, HID, HID, 0},
        {esw2, 64, HID, 0},
        {fxw2, HID, HID, 0}, {fxw3, HID, HID, 0},
        {fyw1, HID, HID, 0}, {fyw2, HID, HID, 0},
        {Wih_f, 384, 768, 0}, {Whh_f, HID, 768, 0},
        {rgw1, HID, 512, 0},  {rgw2, 512, 480, 0},
    };
    const int NW = sizeof(wr) / sizeof(wr[0]);
    for (int i = 0; i < NW; ++i) wr[i].dst = (__bf16*)alloc((size_t)wr[i].K * wr[i].N * 2);
    __bf16 *Wk_b = wr[0].dst, *Wq_b = wr[1].dst, *ekw2b = wr[2].dst, *ekw3b = wr[3].dst,
           *enw2b = wr[4].dst, *enw3b = wr[5].dst, *eqw1b = wr[6].dst, *eqw2b = wr[7].dst,
           *eqw3b = wr[8].dst, *esw2b = wr[9].dst, *fxw2b = wr[10].dst, *fxw3b = wr[11].dst,
           *fyw1b = wr[12].dst, *fyw2b = wr[13].dst, *Wihb = wr[14].dst, *Whhb = wr[15].dst,
           *rgw1b = wr[16].dst, *rgw2b = wr[17].dst;

    // activations / features
    float* kfeat = (float*)alloc((size_t)MBIG * 3 * 4);
    float* nfeat = (float*)alloc((size_t)MBIG * 4 * 4);
    float* sfeat = (float*)alloc((size_t)MS * 4 * 4);
    unsigned char* mask  = (unsigned char*)alloc(MBIG);
    float* dp0   = (float*)alloc((size_t)MFX * 2 * 4);
    unsigned char* mask0 = (unsigned char*)alloc(MFX);

    __bf16* big0 = (__bf16*)alloc((size_t)MBIG * HID * 2);   // embed_k ping / k_e
    __bf16* big1 = (__bf16*)alloc((size_t)MBIG * HID * 2);   // embed_k pong; later split for embed_n
    __bf16* n_a  = big1;                                      // 458752 x 128 scratch
    __bf16* n_b  = big1 + (size_t)MBIG * 128;                 // 458752 x 128 (n_e final)

    __bf16* s1   = (__bf16*)alloc((size_t)MS * 64 * 2);
    __bf16* se_b = (__bf16*)alloc((size_t)MS * HID * 2);
    float*  se_f = (float*)alloc((size_t)MS * HID * 4);
    float*  qe_f = (float*)alloc((size_t)MS * HID * 4);
    float*  sewk = (float*)alloc((size_t)MS * HID * 4);
    float*  ksum = (float*)alloc((size_t)NA * HID * 4);
    float*  mwf  = (float*)alloc((size_t)MS * 4);

    __bf16* fx1  = (__bf16*)alloc((size_t)MFX * HID * 2);
    __bf16* fx2  = (__bf16*)alloc((size_t)MFX * HID * 2);
    float*  fxo  = (float*)alloc((size_t)MFX * HID * 4);

    float*  hf   = (float*)alloc((size_t)NA * HID * 4);
    __bf16* hb   = (__bf16*)alloc((size_t)NA * HID * 2);
    __bf16* q1b  = (__bf16*)alloc((size_t)NA * HID * 2);
    __bf16* q2b  = (__bf16*)alloc((size_t)NA * HID * 2);
    float*  qfb  = (float*)alloc((size_t)NA * HID * 4);
    __bf16* xin  = (__bf16*)alloc((size_t)NA * 384 * 2);
    float*  gi_f = (float*)alloc((size_t)NA * 768 * 4);
    float*  gh_f = (float*)alloc((size_t)NA * 768 * 4);
    __bf16* hy1  = (__bf16*)alloc((size_t)NA * HID * 2);
    __bf16* hy2  = (__bf16*)alloc((size_t)NA * HID * 2);
    __bf16* rg1b = (__bf16*)alloc((size_t)NA * 512 * 2);
    float*  pred = (float*)alloc((size_t)NA * 480 * 4);

    float* out = (float*)d_out;

    // --- weight packing ------------------------------------------------------
    for (int i = 0; i < NW; ++i)
        pack_w<<<grid1((long long)wr[i].K * wr[i].N, 256), 256, 0, stream>>>(
            wr[i].src, wr[i].dst, wr[i].K, wr[i].N);

    // --- preprocess ----------------------------------------------------------
    preprocess<<<grid1(MBIG, 256), 256, 0, stream>>>(x, nb, kfeat, nfeat, sfeat, mask, dp0, mask0);

    auto gemm = [&](const __bf16* X, const __bf16* W, const float* b,
                    __bf16* Yb, float* Yf, int M, int K, int N, int act) {
        dim3 g(M / 32, (N + 63) / 64);
        wmma_gemm<<<g, 256, 0, stream>>>(X, W, b, Yb, Yf, M, K, N, act);
    };

    // --- embed_k: [3,256,256,256], relu6 on first two -----------------------
    dense_small<<<grid1((long long)MBIG * HID, 256), 256, 0, stream>>>(kfeat, ekw1, ekb1, big0, MBIG, 3, HID, 1);
    gemm(big0, ekw2b, ekb2, big1, nullptr, MBIG, HID, HID, 1);
    gemm(big1, ekw3b, ekb3, big0, nullptr, MBIG, HID, HID, 0);    // k_e = big0

    // --- embed_n: [4,64,128,128] --------------------------------------------
    dense_small<<<grid1((long long)MBIG * 64, 256), 256, 0, stream>>>(nfeat, enw1, enb1, n_a, MBIG, 4, 64, 1);
    gemm(n_a, enw2b, enb2, n_a + (size_t)MBIG * 64, nullptr, MBIG, 64, 128, 1);
    gemm(n_a + (size_t)MBIG * 64, enw3b, enb3, n_b, nullptr, MBIG, 128, 128, 0);  // n_e = n_b

    // --- embed_s: [4,64,256] -------------------------------------------------
    dense_small<<<grid1((long long)MS * 64, 256), 256, 0, stream>>>(sfeat, esw1, esb1, s1, MS, 4, 64, 1);
    gemm(s1, esw2b, esb2, se_b, se_f, MS, 64, HID, 0);

    // --- meta weights --------------------------------------------------------
    gemm(se_b, Wq_b, nullptr, nullptr, qe_f, MS, HID, HID, 0);
    gemm(se_b, Wk_b, nullptr, nullptr, sewk, MS, HID, HID, 0);
    ksum_kernel<<<grid1(NA * HID, 256), 256, 0, stream>>>(sewk, ksum);
    metaw_kernel<<<grid1(MS, 256), 256, 0, stream>>>(qe_f, ksum, mwf);

    // --- h0 = masked sum of fx_init(dp0) ------------------------------------
    dense_small<<<grid1((long long)MFX * HID, 256), 256, 0, stream>>>(dp0, fxw1, fxb1, fx1, MFX, 2, HID, 1);
    gemm(fx1, fxw2b, fxb2, fx2, nullptr, MFX, HID, HID, 1);
    gemm(fx2, fxw3b, fxb3, nullptr, fxo, MFX, HID, HID, 0);
    masked_sum<<<grid1(NA * HID, 256), 256, 0, stream>>>(fxo, mask0, hf, hb);

    // --- GRU scan: 7 launch-chained steps ------------------------------------
    for (int t = 0; t < LM1; ++t) {
        // q = embed_q(h): [256,256,256,256] relu6 on first two
        gemm(hb, eqw1b, eqb1, q1b, nullptr, NA, HID, HID, 1);
        gemm(q1b, eqw2b, eqb2, q2b, nullptr, NA, HID, HID, 1);
        gemm(q2b, eqw3b, eqb3, nullptr, qfb, NA, HID, HID, 0);
        // attention + xin
        attn_kernel<<<dim3(NA / 8), 256, 0, stream>>>(t, big0, qfb, mask, n_b, se_f, mwf, xin);
        // gates
        gemm(xin, Wihb, bih, nullptr, gi_f, NA, 384, 768, 0);
        gemm(hb,  Whhb, bhh, nullptr, gh_f, NA, HID, 768, 0);
        gru_update<<<grid1(NA * HID, 256), 256, 0, stream>>>(gi_f, gh_f, hf, hb);
    }

    // --- heads ---------------------------------------------------------------
    gemm(hb, fyw1b, fyb1, hy1, nullptr, NA, HID, HID, 1);
    gemm(hy1, fyw2b, fyb2, hy2, nullptr, NA, HID, HID, 0);
    gemm(hy2, rgw1b, rgb1, rg1b, nullptr, NA, HID, 512, 1);
    gemm(rg1b, rgw2b, rgb2, nullptr, pred, NA, 512, 480, 0);

    final_kernel<<<grid1(NA * 20 * 2, 256), 256, 0, stream>>>(pred, x, out);
}
